// MiniMaxText01LinearAttention_61555471286892
// MI455X (gfx1250) — matrix operand order repaired and checked by reading, not compiled, not run on hardware
//
// MiniMax Text01 linear-attention layer for MI455X (gfx1250, wave32, WMMA).
// f32 inputs converted once to bf16 (weights then live in the 192MB L2); all
// GEMMs + attention matmuls via v_wmma_f32_16x16x32_bf16 with f32 accumulate.
// GEMM operand tiles staged to LDS via the Tensor Data Mover when available
// (tensor_load_to_lds + s_wait_tensorcnt), with a global_load_b128 fallback.
#include <hip/hip_runtime.h>
#include <hip/hip_bf16.h>
#include <math.h>

typedef __attribute__((ext_vector_type(8)))  __bf16 v8bf;
typedef __attribute__((ext_vector_type(16))) __bf16 v16bf;
typedef __attribute__((ext_vector_type(8)))  float  v8f;
typedef __attribute__((ext_vector_type(4)))  unsigned int v4u;
typedef __attribute__((ext_vector_type(8)))  int v8i;
typedef __attribute__((ext_vector_type(4)))  int v4i;

#if defined(__has_builtin)
#if __has_builtin(__builtin_amdgcn_tensor_load_to_lds) && \
    __has_builtin(__builtin_amdgcn_s_wait_tensorcnt)
#define HAVE_TDM 1
#endif
#endif
#if defined(__has_include)
#if __has_include(<hip/amd_detail/amd_gfx1250_TDM.h>)
#define TDM_ARITY6 1
#endif
#endif

static __device__ __forceinline__ float bf2f(unsigned short u) {
  unsigned x = ((unsigned)u) << 16;
  return __builtin_bit_cast(float, x);
}
static __device__ __forceinline__ unsigned short f2bf(float f) {
  unsigned u = __builtin_bit_cast(unsigned, f);
  unsigned r = u + 0x7FFFu + ((u >> 16) & 1u);   // round-to-nearest-even
  return (unsigned short)(r >> 16);
}
static __device__ __forceinline__ __bf16 us2bf(unsigned short u) {
  return __builtin_bit_cast(__bf16, u);
}
static __device__ __forceinline__ unsigned short bf2us(__bf16 b) {
  return __builtin_bit_cast(unsigned short, b);
}
static __device__ __forceinline__ v16bf cat16(v8bf lo, v8bf hi) {
  return __builtin_shufflevector(lo, hi, 0,1,2,3,4,5,6,7,8,9,10,11,12,13,14,15);
}
static __device__ __forceinline__ v8f vzero8() {
  v8f z;
#pragma unroll
  for (int e = 0; e < 8; e++) z[e] = 0.f;
  return z;
}

#if HAVE_TDM
// Issue one TDM 2D tile load: 128 rows x 64 bf16 cols, row stride K elements,
// LDS padded +4 DWORDs per 32 DWORDs (-> 72-element LDS rows).
static __device__ __forceinline__ void tdm_load_tile_2d(
    const unsigned short* gsrc, unsigned lds_byte, int K) {
  unsigned long long ga = (unsigned long long)(const void*)gsrc;
  v4u g0;
  g0[0] = 1u;                                             // count=1 (valid)
  g0[1] = lds_byte;                                       // lds_addr
  g0[2] = (unsigned)(ga & 0xFFFFFFFFu);                   // global_addr[31:0]
  g0[3] = (unsigned)((ga >> 32) & 0x01FFFFFFu) | (2u << 30); // addr[56:32]|type=2
  v8i g1;
  // data_size=1(2B) | pad_enable | pad_interval=4(32 DW) | pad_amount=3(4 DW)
  g1[0] = (int)((1u << 16) | (1u << 20) | (4u << 22) | (3u << 25));
  g1[1] = (int)(((unsigned)K & 0xFFFFu) << 16);           // tensor_dim0 lo16
  g1[2] = (int)((((unsigned)K >> 16) & 0xFFFFu) | (16384u << 16)); // dim0 hi|dim1 lo
  g1[3] = (int)(64u << 16);                               // dim1 hi=0 | tile_dim0=64
  g1[4] = 128;                                            // tile_dim1=128, tile_dim2=0
  g1[5] = K;                                              // tensor_dim0_stride lo32
  g1[6] = 0;
  g1[7] = 0;
  v4i g2, g3;
#pragma unroll
  for (int e = 0; e < 4; e++) { g2[e] = 0; g3[e] = 0; }
#if TDM_ARITY6
  v8i g4;
#pragma unroll
  for (int e = 0; e < 8; e++) g4[e] = 0;
  __builtin_amdgcn_tensor_load_to_lds(g0, g1, g2, g3, g4, 0);
#else
  __builtin_amdgcn_tensor_load_to_lds(g0, g1, g2, g3, 0);
#endif
}
#endif

// ---------------------------------------------------------------------------
// f32 -> bf16 bulk convert
// ---------------------------------------------------------------------------
__global__ void cvt_bf16_kernel(const float* __restrict__ in,
                                unsigned short* __restrict__ out, size_t n) {
  size_t i = (size_t)blockIdx.x * blockDim.x + threadIdx.x;
  size_t stride = (size_t)gridDim.x * blockDim.x;
  for (; i < n; i += stride) out[i] = f2bf(in[i]);
}

// ---------------------------------------------------------------------------
// bf16 WMMA GEMM: C[m,n] = act( sum_k A[m,k]*B[n,k] )   (both row-major)
// 128x128 block tile, K-step 64, 8 waves each computing 32x64 via 2x4 WMMA tiles.
// MODE 0: silu -> bf16 | MODE 1: sigmoid*hidden*rstd*normw -> bf16 | MODE 2: f32
// ---------------------------------------------------------------------------
template <int MODE>
__global__ __launch_bounds__(256) void gemm_bf16_kernel(
    const unsigned short* __restrict__ A, const unsigned short* __restrict__ B,
    int M, int N, int K,
    unsigned short* __restrict__ out_bf, float* __restrict__ out_f,
    const float* __restrict__ hidden, const float* __restrict__ rstd,
    const float* __restrict__ normw) {
  constexpr int TM = 128, TN = 128, TK = 64, LDT = TK + 8;
  __shared__ union {
    unsigned short ab[2][TM * LDT];   // staged A (ab[0]) and B (ab[1]) tiles
    float c[8][16 * 64];              // per-wave epilogue bounce (aliased)
  } sm;
  const int tid = threadIdx.x, lane = tid & 31, wave = tid >> 5;
  const int wm = wave & 3, wn = wave >> 2;
  const int m0 = blockIdx.y * TM, n0 = blockIdx.x * TN;
  const int coll = lane & 15, rowo = (lane >> 4) * 8;

  v8f acc[2][4];
#pragma unroll
  for (int i = 0; i < 2; i++)
#pragma unroll
    for (int j = 0; j < 4; j++) acc[i][j] = vzero8();

#if HAVE_TDM
  const unsigned ldsA = (unsigned)(size_t)(void*)sm.ab[0];
  const unsigned ldsB = (unsigned)(size_t)(void*)sm.ab[1];
#endif

  for (int k0 = 0; k0 < K; k0 += TK) {
    __syncthreads();
#if HAVE_TDM
    if (wave == 0) {
      tdm_load_tile_2d(A + (size_t)m0 * K + k0, ldsA, K);
      tdm_load_tile_2d(B + (size_t)n0 * K + k0, ldsB, K);
      __builtin_amdgcn_s_wait_tensorcnt(0);
    }
#else
#pragma unroll
    for (int v = 0; v < 4; v++) {
      int idx = tid + v * 256;            // 1024 16B vectors per operand tile
      int r = idx >> 3, c = (idx & 7) * 8;
      *reinterpret_cast<v8bf*>(&sm.ab[0][r * LDT + c]) =
          *reinterpret_cast<const v8bf*>(A + (size_t)(m0 + r) * K + k0 + c);
      *reinterpret_cast<v8bf*>(&sm.ab[1][r * LDT + c]) =
          *reinterpret_cast<const v8bf*>(B + (size_t)(n0 + r) * K + k0 + c);
      if (k0 + TK < K) {
        __builtin_prefetch(A + (size_t)(m0 + r) * K + k0 + TK + c, 0, 1);
        __builtin_prefetch(B + (size_t)(n0 + r) * K + k0 + TK + c, 0, 1);
      }
    }
#endif
    __syncthreads();
#pragma unroll
    for (int ks = 0; ks < TK; ks += 32) {
      v16bf af[2], bb[4];
      const int akb = ks + ((lane >> 4) ? 8 : 0);
#pragma unroll
      for (int i = 0; i < 2; i++) {
        int r = wm * 32 + i * 16 + coll;
        af[i] = cat16(*reinterpret_cast<const v8bf*>(&sm.ab[0][r * LDT + akb]),
                      *reinterpret_cast<const v8bf*>(&sm.ab[0][r * LDT + akb + 16]));
      }
      const int bkb = ks + ((lane >> 4) ? 8 : 0);
#pragma unroll
      for (int j = 0; j < 4; j++) {
        int r = wn * 64 + j * 16 + coll;
        bb[j] = cat16(*reinterpret_cast<const v8bf*>(&sm.ab[1][r * LDT + bkb]),
                      *reinterpret_cast<const v8bf*>(&sm.ab[1][r * LDT + bkb + 16]));
      }
#pragma unroll
      for (int i = 0; i < 2; i++)
#pragma unroll
        for (int j = 0; j < 4; j++)
          acc[i][j] = __builtin_amdgcn_wmma_f32_16x16x32_bf16(
              false, af[i], false, bb[j], (short)0, acc[i][j], false, false);
    }
  }

  // Epilogue: bounce each 16x64 accumulator chunk through per-wave LDS so the
  // global writes become per-lane 128B-contiguous runs.
  __syncthreads();   // done reading staged tiles; union now used as bounce
  const int lrow = lane >> 1, lcb = (lane & 1) * 32;
#pragma unroll
  for (int i = 0; i < 2; i++) {
    float* cw = sm.c[wave];
#pragma unroll
    for (int j = 0; j < 4; j++)
#pragma unroll
      for (int r = 0; r < 8; r++)
        cw[(rowo + r) * 64 + j * 16 + coll] = acc[i][j][r];
    // intra-wave DS ops are in-order; no cross-wave sharing of this region
    const int m = m0 + wm * 32 + i * 16 + lrow;
    const int nb = n0 + wn * 64 + lcb;
#pragma unroll
    for (int c = 0; c < 32; c++) {
      float v = cw[lrow * 64 + lcb + c];
      const int n = nb + c;
      if (MODE == 0) {
        float sv = v / (1.f + __expf(-v));
        out_bf[(size_t)m * N + n] = f2bf(sv);
      } else if (MODE == 1) {
        float g = 1.f / (1.f + __expf(-v));
        float res = g * hidden[(size_t)m * N + n] * rstd[m] * normw[n];
        out_bf[(size_t)m * N + n] = f2bf(res);
      } else {
        out_f[(size_t)m * N + n] = v;
      }
    }
    __syncthreads();
  }
}

// ---------------------------------------------------------------------------
// Lightning attention: one workgroup per head, sequential over 16 blocks of 256.
// Dynamic LDS (320KB = full WGP budget):
//   kvT [128e][128d] bf16 (state, transposed)  @ 0      (32KB)
//   qkb [256m][256n] bf16 (masked q@k^T)       @ 32KB   (128KB)
//   vt  [128e][256n] bf16 (v transposed)       @ 160KB  (64KB)
//   ktd [128d][256n] bf16 ((k*kdec)^T)         @ 224KB  (64KB)
//   ob  per-wave f32 bounce                    @ 288KB  (32KB)
// ---------------------------------------------------------------------------
__global__ __launch_bounds__(256) void attn_kernel(
    const unsigned short* __restrict__ qkv,  // SEQ x 12288 bf16
    const float* __restrict__ kvc,           // NH x 128 x 128 f32
    const float* __restrict__ slope,         // NH
    float* __restrict__ hidden) {            // SEQ x 4096 f32
  extern __shared__ unsigned char smem_raw[];
  unsigned short* kvT = (unsigned short*)smem_raw;
  unsigned short* qkb = (unsigned short*)(smem_raw + 32 * 1024);
  unsigned short* vt  = (unsigned short*)(smem_raw + 160 * 1024);
  unsigned short* ktd = (unsigned short*)(smem_raw + 224 * 1024);
  float*          obp = (float*)(smem_raw + 288 * 1024);
  const int h = blockIdx.x;
  const float s = slope[h];
  const int tid = threadIdx.x, lane = tid & 31, wave = tid >> 5;
  const int coll = lane & 15, rowo = (lane >> 4) * 8;
  const size_t ROW = 12288;
  const size_t qoff = (size_t)h * 384;
  float* ob = obp + wave * 1024;           // 16x64 f32 per wave

  for (int idx = tid; idx < 128 * 128; idx += 256) {
    int e = idx >> 7, d = idx & 127;
    kvT[e * 128 + d] = f2bf(kvc[(size_t)h * 16384 + (size_t)d * 128 + e]);
  }
  __syncthreads();

  const float blkdec = __expf(-s * 256.f);

  for (int blk = 0; blk < 16; blk++) {
    const int row0 = blk * 256;

    // stage v^T and (k * k_dec)^T
    for (int idx = tid; idx < 256 * 128; idx += 256) {
      int n = idx >> 7, d = idx & 127;
      unsigned short kb_ = qkv[(size_t)(row0 + n) * ROW + qoff + 128 + d];
      unsigned short vb_ = qkv[(size_t)(row0 + n) * ROW + qoff + 256 + d];
      vt[d * 256 + n] = vb_;
      float kdec = __expf(-s * (float)(255 - n));
      ktd[d * 256 + n] = f2bf(bf2f(kb_) * kdec);
    }
    __syncthreads();

    // qk = tril_decay(q @ k^T) -> bf16 LDS
    {
      const int mbase = wave * 32;
      for (int i = 0; i < 2; i++) {
        const int mb = mbase + i * 16;
        for (int nt = 0; nt < 16; nt++) {
          const int nb = nt * 16;
          if (nb > mb + 15) {   // fully above diagonal -> zero
#pragma unroll
            for (int r = 0; r < 8; r++) qkb[(mb + rowo + r) * 256 + nb + coll] = 0;
            continue;
          }
          v8f c = vzero8();
#pragma unroll
          for (int kk4 = 0; kk4 < 4; kk4++) {
            const int mrow = mb + coll;
            const unsigned short* qp =
                qkv + (size_t)(row0 + mrow) * ROW + qoff + kk4 * 32 + ((lane >> 4) ? 8 : 0);
            v16bf fa = cat16(*reinterpret_cast<const v8bf*>(qp),
                             *reinterpret_cast<const v8bf*>(qp + 16));
            const int nrow = nb + coll;
            const unsigned short* kp =
                qkv + (size_t)(row0 + nrow) * ROW + qoff + 128 + kk4 * 32 + ((lane >> 4) ? 8 : 0);
            v16bf fb = cat16(*reinterpret_cast<const v8bf*>(kp),
                             *reinterpret_cast<const v8bf*>(kp + 16));
            c = __builtin_amdgcn_wmma_f32_16x16x32_bf16(false, fa, false, fb,
                                                        (short)0, c, false, false);
          }
#pragma unroll
          for (int r = 0; r < 8; r++) {
            const int m = mb + rowo + r, n = nb + coll;
            const int dd = m - n;
            float v = (dd >= 0) ? c[r] * __expf(-s * (float)dd) : 0.f;
            qkb[m * 256 + n] = f2bf(v);
          }
        }
      }
    }
    __syncthreads();

    // o = (q*q_dec) @ kv(old) + qk @ v  -> hidden
    {
      const int mbase = wave * 32;
      for (int half = 0; half < 2; half++) {
        v8f acc[2][4];
#pragma unroll
        for (int i = 0; i < 2; i++)
#pragma unroll
          for (int j = 0; j < 4; j++) acc[i][j] = vzero8();

#pragma unroll
        for (int kk4 = 0; kk4 < 4; kk4++) {          // term 1, K = 128 (d)
          v16bf fa[2];
#pragma unroll
          for (int i = 0; i < 2; i++) {
            const int mrow = mbase + i * 16 + coll;
            const float qd = __expf(-s * (float)(mrow + 1));
            const unsigned short* qp =
                qkv + (size_t)(row0 + mrow) * ROW + qoff + kk4 * 32 + ((lane >> 4) ? 8 : 0);
            v8bf lo = *reinterpret_cast<const v8bf*>(qp);
            v8bf hi = *reinterpret_cast<const v8bf*>(qp + 16);
            v16bf f;
#pragma unroll
            for (int e = 0; e < 8; e++) {
              f[e]     = us2bf(f2bf(bf2f(bf2us(lo[e])) * qd));
              f[e + 8] = us2bf(f2bf(bf2f(bf2us(hi[e])) * qd));
            }
            fa[i] = f;
          }
#pragma unroll
          for (int j = 0; j < 4; j++) {
            const int e = half * 64 + j * 16 + coll;
            const int kb = kk4 * 32 + ((lane >> 4) ? 8 : 0);
            v16bf fb = cat16(*reinterpret_cast<const v8bf*>(&kvT[e * 128 + kb]),
                             *reinterpret_cast<const v8bf*>(&kvT[e * 128 + kb + 16]));
#pragma unroll
            for (int i = 0; i < 2; i++)
              acc[i][j] = __builtin_amdgcn_wmma_f32_16x16x32_bf16(
                  false, fa[i], false, fb, (short)0, acc[i][j], false, false);
          }
        }
#pragma unroll
        for (int kk8 = 0; kk8 < 8; kk8++) {          // term 2, K = 256 (n)
          v16bf fa[2];
#pragma unroll
          for (int i = 0; i < 2; i++) {
            const int mrow = mbase + i * 16 + coll;
            const int kb = kk8 * 32 + ((lane >> 4) ? 8 : 0);
            fa[i] = cat16(*reinterpret_cast<const v8bf*>(&qkb[mrow * 256 + kb]),
                          *reinterpret_cast<const v8bf*>(&qkb[mrow * 256 + kb + 16]));
          }
#pragma unroll
          for (int j = 0; j < 4; j++) {
            const int e = half * 64 + j * 16 + coll;
            const int kb = kk8 * 32 + ((lane >> 4) ? 8 : 0);
            v16bf fb = cat16(*reinterpret_cast<const v8bf*>(&vt[e * 256 + kb]),
                             *reinterpret_cast<const v8bf*>(&vt[e * 256 + kb + 16]));
#pragma unroll
            for (int i = 0; i < 2; i++)
              acc[i][j] = __builtin_amdgcn_wmma_f32_16x16x32_bf16(
                  false, fa[i], false, fb, (short)0, acc[i][j], false, false);
          }
        }
        // store via per-wave LDS bounce -> 128B-contiguous global runs
#pragma unroll
        for (int i = 0; i < 2; i++) {
#pragma unroll
          for (int j = 0; j < 4; j++)
#pragma unroll
            for (int r = 0; r < 8; r++)
              ob[(rowo + r) * 64 + j * 16 + coll] = acc[i][j][r];
          const int mrow = row0 + mbase + i * 16 + (lane >> 1);
          const int cb = (lane & 1) * 32;
          const float* src = ob + (lane >> 1) * 64 + cb;
#pragma unroll
          for (int c = 0; c < 32; c++)
            hidden[(size_t)mrow * 4096 + h * 128 + half * 64 + cb + c] = src[c];
        }
      }
    }
    __syncthreads();

    // kv <- blkdec*kv + (k*kdec)^T @ v  (stored transposed in kvT[e][d])
    {
      const int dbase = wave * 16;
      for (int et = 0; et < 8; et++) {
        v8f c;
#pragma unroll
        for (int r = 0; r < 8; r++) {
          const int e = et * 16 + coll;
          const int d = dbase + rowo + r;
          c[r] = blkdec * bf2f(kvT[e * 128 + d]);
        }
#pragma unroll
        for (int kk8 = 0; kk8 < 8; kk8++) {
          const int drow = dbase + coll;
          const int kba = kk8 * 32 + ((lane >> 4) ? 8 : 0);
          v16bf fa = cat16(*reinterpret_cast<const v8bf*>(&ktd[drow * 256 + kba]),
                           *reinterpret_cast<const v8bf*>(&ktd[drow * 256 + kba + 16]));
          const int e = et * 16 + coll;
          const int kbb = kk8 * 32 + ((lane >> 4) ? 8 : 0);
          v16bf fb = cat16(*reinterpret_cast<const v8bf*>(&vt[e * 256 + kbb]),
                           *reinterpret_cast<const v8bf*>(&vt[e * 256 + kbb + 16]));
          c = __builtin_amdgcn_wmma_f32_16x16x32_bf16(false, fa, false, fb,
                                                      (short)0, c, false, false);
        }
#pragma unroll
        for (int r = 0; r < 8; r++) {
          const int e = et * 16 + coll;
          const int d = dbase + rowo + r;
          kvT[e * 128 + d] = f2bf(c[r]);
        }
      }
    }
    __syncthreads();
  }
}

// ---------------------------------------------------------------------------
// Per-row rstd = rsqrt(mean(h^2)+eps)
// ---------------------------------------------------------------------------
__global__ __launch_bounds__(256) void rms_kernel(const float* __restrict__ hidden,
                                                  float* __restrict__ rstd) {
  __shared__ float red[256];
  const int row = blockIdx.x;
  float sum = 0.f;
  for (int c = threadIdx.x; c < 4096; c += 256) {
    float v = hidden[(size_t)row * 4096 + c];
    sum += v * v;
  }
  red[threadIdx.x] = sum;
  __syncthreads();
  for (int s = 128; s > 0; s >>= 1) {
    if ((int)threadIdx.x < s) red[threadIdx.x] += red[threadIdx.x + s];
    __syncthreads();
  }
  if (threadIdx.x == 0) rstd[row] = rsqrtf(red[0] / 4096.f + 1e-5f);
}

// ---------------------------------------------------------------------------
extern "C" void kernel_launch(void* const* d_in, const int* in_sizes, int n_in,
                              void* d_out, int out_size, void* d_ws, size_t ws_size,
                              hipStream_t stream) {
  (void)in_sizes; (void)n_in; (void)out_size; (void)ws_size;
  const float* x      = (const float*)d_in[0];
  const float* w_qkv  = (const float*)d_in[1];
  const float* w_gate = (const float*)d_in[2];
  const float* w_out  = (const float*)d_in[3];
  const float* normw  = (const float*)d_in[4];
  const float* kvc    = (const float*)d_in[5];
  const float* slope  = (const float*)d_in[6];
  float* out = (float*)d_out;

  const size_t SEQ = 4096, H = 4096, NQKV = 12288;

  char* ws = (char*)d_ws;
  unsigned short* x_bf    = (unsigned short*)ws;  ws += SEQ * H * 2;
  unsigned short* wqkv_bf = (unsigned short*)ws;  ws += NQKV * H * 2;
  unsigned short* wgate_bf= (unsigned short*)ws;  ws += H * H * 2;
  unsigned short* wout_bf = (unsigned short*)ws;  ws += H * H * 2;
  unsigned short* qkv_bf  = (unsigned short*)ws;  ws += SEQ * NQKV * 2;
  float*          hidden  = (float*)ws;           ws += SEQ * H * 4;
  unsigned short* gn_bf   = (unsigned short*)ws;  ws += SEQ * H * 2;
  float*          rstd    = (float*)ws;           ws += SEQ * 4;

  cvt_bf16_kernel<<<2048, 256, 0, stream>>>(x, x_bf, SEQ * H);
  cvt_bf16_kernel<<<2048, 256, 0, stream>>>(w_qkv, wqkv_bf, NQKV * H);
  cvt_bf16_kernel<<<2048, 256, 0, stream>>>(w_gate, wgate_bf, H * H);
  cvt_bf16_kernel<<<2048, 256, 0, stream>>>(w_out, wout_bf, H * H);

  // qkv = silu(x @ w_qkv^T)  -> bf16
  gemm_bf16_kernel<0><<<dim3(96, 32), 256, 0, stream>>>(
      x_bf, wqkv_bf, 4096, 12288, 4096, qkv_bf, nullptr, nullptr, nullptr, nullptr);

  // lightning attention, one workgroup per head, full 320KB dynamic LDS
  const int attn_lds = 320 * 1024;
  hipFuncSetAttribute(reinterpret_cast<const void*>(attn_kernel),
                      hipFuncAttributeMaxDynamicSharedMemorySize, attn_lds);
  attn_kernel<<<32, 256, attn_lds, stream>>>(qkv_bf, kvc, slope, hidden);

  rms_kernel<<<4096, 256, 0, stream>>>(hidden, rstd);

  // gn = sigmoid(x @ w_gate^T) * hidden * rstd * norm_w -> bf16
  gemm_bf16_kernel<1><<<dim3(32, 32), 256, 0, stream>>>(
      x_bf, wgate_bf, 4096, 4096, 4096, gn_bf, nullptr, hidden, rstd, normw);

  // out = gn @ w_out^T -> f32
  gemm_bf16_kernel<2><<<dim3(32, 32), 256, 0, stream>>>(
      gn_bf, wout_bf, 4096, 4096, 4096, nullptr, out, nullptr, nullptr, nullptr);
}
